// context_mapping_24240795418922
// MI455X (gfx1250) — matrix operand order repaired and checked by reading, not compiled, not run on hardware
//
#include <hip/hip_runtime.h>
#include <hip/hip_bf16.h>

// CDNA5 / gfx1250 context-mapping kernel.
// Per-pixel MLP (64->32->16->8->1) done with v_wmma_f32_16x16x32_bf16,
// D = W x X orientation so inter-layer relayout is a single lane xor-16
// exchange, implemented with v_permlanex16_b32 (pure VALU, no LDS traffic).
// Distance-matrix branch (period 4x4) hoisted out of the pixel loop.

typedef __attribute__((ext_vector_type(16))) __bf16 v16bf;
typedef __attribute__((ext_vector_type(8)))  float  v8f;

#define LRELU(x) fmaxf((x), 0.01f * (x))   // slope 0.01 > 0 => lrelu == max(x, 0.01x)

// lane i <-> lane i^16 exchange via v_permlanex16_b32 with identity selects:
// lo-half lane i reads hi-half lane i, hi-half lane i reads lo-half lane i.
__device__ __forceinline__ float xor16(float v) {
    unsigned int u = __builtin_bit_cast(unsigned int, v);
    unsigned int r = __builtin_amdgcn_permlanex16(u, u, 0x76543210u, 0xfedcba98u,
                                                  false, false);
    return __builtin_bit_cast(float, r);
}

#define WMMA_BF16(A, B, C) \
    __builtin_amdgcn_wmma_f32_16x16x32_bf16(false, (A), false, (B), (short)0, (C), false, false)

__global__ __launch_bounds__(256)
void ctxmap_wmma_kernel(const float* __restrict__ lr, const float* __restrict__ hr,
                        const float* __restrict__ w0, const float* __restrict__ w1,
                        const float* __restrict__ w2, const float* __restrict__ w3,
                        const float* __restrict__ s0, const float* __restrict__ s1,
                        const float* __restrict__ s2, const float* __restrict__ wf,
                        float* __restrict__ out)
{
    const int W = 1024, h_lr = 256;
    const int y    = blockIdx.x;           // one hr row per block (1024 blocks)
    const int wave = threadIdx.x >> 5;     // 8 waves per block
    const int lane = threadIdx.x & 31;
    const int n    = lane & 15;            // pixel slot within 16-wide tile
    const bool hi  = lane >= 16;
    const int kofs = hi ? 8 : 0;           // A/B element->K mapping half-wave offset

    // ---------------- weight tiles in WMMA A layout (once per wave) -------------
    // A (16-bit, 16xK=32), contiguous K order chosen here and matched in B: lane L<16 holds K {0..15}; lane L+16 holds {16..31}.
    v16bf Aw0[2][2];                       // [oc-half][K-chunk of rep(64)]
    #pragma unroll
    for (int hh = 0; hh < 2; ++hh) {
        const int oc = hh * 16 + n;
        #pragma unroll
        for (int c = 0; c < 2; ++c) {
            #pragma unroll
            for (int e = 0; e < 16; ++e) {
                const int k = e + 2 * kofs;
                Aw0[hh][c][e] = (__bf16)w0[oc * 64 + c * 32 + k];
            }
        }
    }
    v16bf Aw1;                              // w1: 16 oc x 32 ic
    #pragma unroll
    for (int e = 0; e < 16; ++e) {
        const int k = e + 2 * kofs;
        Aw1[e] = (__bf16)w1[n * 32 + k];
    }
    v16bf Aw2;                              // w2: 8 oc x 16 ic, zero-padded to 16x32
    #pragma unroll
    for (int e = 0; e < 16; ++e) {
        const int k = e + 2 * kofs;
        const float v = (n < 8 && k < 16) ? w2[n * 16 + k] : 0.f;
        Aw2[e] = (__bf16)v;
    }
    float w3r[8];
    #pragma unroll
    for (int c = 0; c < 8; ++c) w3r[c] = w3[c];
    const float awf0 = fabsf(wf[0]), awf1 = fabsf(wf[1]);

    // ------------- distance-matrix branch: constant per (lane, row) -------------
    // dm values are 4x4-periodic; x0 is a multiple of 16 so (x & 3) == (n & 3).
    const float xv[4] = {-2.f, -1.f, 1.f, 2.f};
    const float d0 = xv[n & 3];
    const float d1 = xv[y & 3];
    const float d2 = sqrtf(d0 * d0 + d1 * d1);
    const float o0 = LRELU(s0[0] * d0 + s0[1] * d1 + s0[2] * d2);
    const float o1 = LRELU(s0[3] * d0 + s0[4] * d1 + s0[5] * d2);
    const float o2 = LRELU(s0[6] * d0 + s0[7] * d1 + s0[8] * d2);
    const float p0 = LRELU(s1[0] * o0 + s1[1] * o1 + s1[2] * o2);
    const float p1 = LRELU(s1[3] * o0 + s1[4] * o1 + s1[5] * o2);
    const float w2map = LRELU(s2[0] * p0 + s2[1] * p1);

    // lanes 0-15 fetch lr (nearest-upsampled), lanes 16-31 fetch hr, same pixel n
    const float* src      = hi ? hr : lr;
    const long   chstride = hi ? (long)(W * W) : (long)(h_lr * h_lr);

    const v8f z = {0.f, 0.f, 0.f, 0.f, 0.f, 0.f, 0.f, 0.f};

    // ------------------------ 8 tiles of 16 pixels per wave ----------------------
    for (int t = 0; t < 8; ++t) {
        const int  x0   = (wave * 8 + t) * 16;
        const int  px   = x0 + n;
        const long base = hi ? (long)y * W + px
                             : (long)(y >> 2) * h_lr + (px >> 2);

        float m[16];
        #pragma unroll
        for (int ch = 0; ch < 16; ++ch) m[ch] = src[ch * chstride + base];

        // B (K=32 x 16px): chunk0 = [lr | hr], chunk1 = [lr*hr | (lr-hr)^2]
        v16bf B0, B1;
        #pragma unroll
        for (int e = 0; e < 16; ++e) {
            const float o = xor16(m[e]);                    // lr <-> hr exchange
            B0[e] = (__bf16)m[e];
            const float v = hi ? (o - m[e]) * (o - m[e])    // (lr - hr)^2
                               : m[e] * o;                  // lr * hr
            B1[e] = (__bf16)v;
        }

        // layer 0: 64 -> 32
        v8f acc0 = z, acc1 = z;
        acc0 = WMMA_BF16(Aw0[0][0], B0, acc0);
        acc0 = WMMA_BF16(Aw0[0][1], B1, acc0);
        acc1 = WMMA_BF16(Aw0[1][0], B0, acc1);
        acc1 = WMMA_BF16(Aw0[1][1], B1, acc1);

        // relayout hidden(32ch) into B for layer 1 via one xor-16 exchange
        v16bf Bh;
        #pragma unroll
        for (int r = 0; r < 8; ++r) {
            const float a0  = LRELU(acc0[r]);
            const float a1  = LRELU(acc1[r]);
            const float a0x = xor16(a0);
            const float a1x = xor16(a1);
            Bh[r]     = (__bf16)(hi ? a1x : a0);   // ch r    | ch 16+r
            Bh[r + 8] = (__bf16)(hi ? a1  : a0x);  // ch 8+r  | ch 24+r
        }
        v8f accB = z;
        accB = WMMA_BF16(Aw1, Bh, accB);           // layer 1: 32 -> 16

        // relayout hidden(16ch), zero-pad K to 32, for layer 2
        v16bf Bh2;
        #pragma unroll
        for (int r = 0; r < 8; ++r) {
            const float h1  = LRELU(accB[r]);
            const float h1x = xor16(h1);
            Bh2[r]     = (__bf16)(hi ? 0.f : h1);
            Bh2[r + 8] = (__bf16)(hi ? 0.f : h1x);
        }
        v8f accC = z;
        accC = WMMA_BF16(Aw2, Bh2, accC);          // layer 2: 16 -> 8

        // layer 3 (8 -> 1) in VALU, then fuse with distance branch
        float sacc = 0.f;
        #pragma unroll
        for (int c = 0; c < 8; ++c) sacc += w3r[c] * LRELU(accC[c]);
        const float w1map   = LRELU(sacc);
        const float mapping = LRELU(awf0 * w1map + awf1 * w2map);

        if (!hi) out[(long)y * W + px] = mapping;  // 64B contiguous store per wave
    }
}

extern "C" void kernel_launch(void* const* d_in, const int* in_sizes, int n_in,
                              void* d_out, int out_size, void* d_ws, size_t ws_size,
                              hipStream_t stream) {
    const float* lr = (const float*)d_in[0];
    const float* hr = (const float*)d_in[1];
    const float* w0 = (const float*)d_in[2];
    const float* w1 = (const float*)d_in[3];
    const float* w2 = (const float*)d_in[4];
    const float* w3 = (const float*)d_in[5];
    const float* s0 = (const float*)d_in[6];
    const float* s1 = (const float*)d_in[7];
    const float* s2 = (const float*)d_in[8];
    const float* wf = (const float*)d_in[9];
    float* outp = (float*)d_out;

    dim3 grid(1024);   // one hr row per block
    dim3 block(256);   // 8 wave32s; each wave: 8 tiles x 16 px = 128 px of the row
    ctxmap_wmma_kernel<<<grid, block, 0, stream>>>(lr, hr, w0, w1, w2, w3,
                                                   s0, s1, s2, wf, outp);
}
